// EncoderWithPositionalAttentionLayer_24807731102179
// MI455X (gfx1250) — matrix operand order repaired and checked by reading, not compiled, not run on hardware
//
#include <hip/hip_runtime.h>
#include <hip/hip_bf16.h>
#include <cstdint>

// ---------------------------------------------------------------------------
// EncoderWithPositionalAttentionLayer for MI455X (gfx1250, wave32, WMMA+TDM)
//
// Algebraic restructuring vs reference:
//   * rel_enc gather has only 201 distinct rows -> kr_small [201,512] GEMM
//   * biasprod -> per-head [T,64]x[64,208] WMMA GEMM into lookup table p
// Dense GEMMs: v_wmma_f32_16x16x32_bf16, weight strips staged into LDS by the
// Tensor Data Mover (double-buffered, TENSORcnt-synced) and shared by all 8
// waves of the workgroup. ~57 GFLOP total, activations L2-resident.
// ---------------------------------------------------------------------------

constexpr int kB   = 8;
constexpr int kT   = 512;
constexpr int kD   = 512;
constexpr int kH   = 8;
constexpr int kDim = 64;
constexpr int kHid = 2048;
constexpr int kL   = 100;
constexpr int kR   = 201;   // 2L+1 distinct relative offsets
constexpr int kRP  = 208;   // padded to multiple of 16 for WMMA N-tiles
constexpr int kBT  = kB * kT;
constexpr int kBH  = kB * kH;

typedef __attribute__((ext_vector_type(16))) __bf16 bf16x16;
typedef __attribute__((ext_vector_type(8)))  float  f32x8;
typedef __attribute__((ext_vector_type(4)))  unsigned int u32x4;
typedef __attribute__((ext_vector_type(8)))  unsigned int u32x8;

union BFrag { bf16x16 v; uint4 u[2]; };

__device__ __forceinline__ unsigned short f2bf(float f) {
  unsigned int u = __float_as_uint(f);
  unsigned int r = u + 0x7FFFu + ((u >> 16) & 1u);   // round-to-nearest-even
  return (unsigned short)(r >> 16);
}
__device__ __forceinline__ float bf2f(unsigned short h) {
  return __uint_as_float((unsigned int)h << 16);
}

// ---------------------------------------------------------------------------
// Weight convert + transpose: src fp32 [K,N] row-major -> dst bf16 [N,K]
// ---------------------------------------------------------------------------
__global__ void cvtT_kernel(const float* __restrict__ src,
                            unsigned short* __restrict__ dst, int Kd, int Nd) {
  long idx = (long)blockIdx.x * blockDim.x + threadIdx.x;
  if (idx >= (long)Kd * Nd) return;
  int n = (int)(idx % Nd), k = (int)(idx / Nd);
  dst[(long)n * Kd + k] = f2bf(src[idx]);
}

// ---------------------------------------------------------------------------
// LayerNorm (row = 512) -> bf16. One wave per row, shuffle reductions.
// ---------------------------------------------------------------------------
__global__ void ln_kernel(const float* __restrict__ x, const float* __restrict__ g,
                          const float* __restrict__ b, unsigned short* __restrict__ out) {
  int row  = blockIdx.x * (blockDim.x >> 5) + (threadIdx.x >> 5);
  int lane = threadIdx.x & 31;
  if (row >= kBT) return;
  const float* xr = x + (long)row * kD;
  float v[16], s = 0.f, ss = 0.f;
#pragma unroll
  for (int i = 0; i < 16; ++i) { float t = xr[lane + 32 * i]; v[i] = t; s += t; ss += t * t; }
#pragma unroll
  for (int m = 16; m >= 1; m >>= 1) { s += __shfl_xor(s, m, 32); ss += __shfl_xor(ss, m, 32); }
  float mean = s * (1.0f / kD);
  float var  = ss * (1.0f / kD) - mean * mean;
  float inv  = rsqrtf(var + 1e-3f);
#pragma unroll
  for (int i = 0; i < 16; ++i) {
    int c = lane + 32 * i;
    out[(long)row * kD + c] = f2bf((v[i] - mean) * inv * g[c] + b[c]);
  }
}

// ---------------------------------------------------------------------------
// Epilogue IDs shared by both GEMM kernels
// ---------------------------------------------------------------------------
enum { EPI_RELU_BF16 = 0, EPI_HEADS = 1, EPI_KVT = 2, EPI_F32 = 3,
       EPI_ATTNOUT = 4, EPI_OUT_RES = 5 };

template <int EPI>
__device__ __forceinline__ void gemm_epilogue(float c, int i, int n, int N, int z,
                                              const float* __restrict__ bias,
                                              const float* __restrict__ residual,
                                              void* __restrict__ out, int M) {
  if constexpr (EPI == EPI_RELU_BF16) {
    c += bias[n]; c = fmaxf(c, 0.f);
    ((unsigned short*)out)[(long)i * N + n] = f2bf(c);
  } else if constexpr (EPI == EPI_HEADS) {          // [B*T,D] -> [B,H,T,DIM] bf16
    c += bias[n];
    int bb = i / kT, tt = i % kT, h = n >> 6, d = n & 63;
    ((unsigned short*)out)[(((long)(bb * kH + h) * kT + tt) << 6) + d] = f2bf(c);
  } else if constexpr (EPI == EPI_KVT) {            // [B*T,D] -> [B,H,DIM,T] bf16
    c += bias[n];
    int bb = i / kT, tt = i % kT, h = n >> 6, d = n & 63;
    ((unsigned short*)out)[((long)(bb * kH + h) * kDim + d) * kT + tt] = f2bf(c);
  } else if constexpr (EPI == EPI_F32) {            // batched fp32 store
    ((float*)out)[((long)z * M + i) * N + n] = c;
  } else if constexpr (EPI == EPI_ATTNOUT) {        // += residual, head-merge
    int bb = z / kH, h = z % kH;
    long idx = ((long)bb * kT + i) * kD + h * kDim + n;   // N == 64
    ((float*)out)[idx] = residual[idx] + c;
  } else if constexpr (EPI == EPI_OUT_RES) {        // final: y1 + dense + bias
    c += bias[n];
    long idx = (long)i * N + n;
    ((float*)out)[idx] = residual[idx] + c;
  }
}

// ---------------------------------------------------------------------------
// Direct-from-L2 WMMA GEMM (batched, small K / small reuse cases).
// Wave computes a 16 x (16*NTILES) strip.
// ---------------------------------------------------------------------------
template <int EPI, int NTILES>
__global__ void gemm16_kernel(const unsigned short* __restrict__ A,
                              const unsigned short* __restrict__ Wt,
                              int M, int N, int K,
                              long Abatch, long Wbatch, int WbatchMod,
                              const float* __restrict__ bias,
                              const float* __restrict__ residual,
                              void* __restrict__ out) {
  const int lane = threadIdx.x & 31;
  const int wid  = (blockIdx.x * blockDim.x + threadIdx.x) >> 5;
  const int wstride = (N >> 4) / NTILES;
  const int iw = wid / wstride, jw = wid % wstride;
  const int i0 = iw * 16;
  if (i0 >= M) return;
  const int n0 = jw * (16 * NTILES);
  const int z  = blockIdx.z;

  const unsigned short* Ab = A + (long)z * Abatch;
  const unsigned short* Wb = Wt + (long)(WbatchMod ? (z % WbatchMod) : z) * Wbatch;

  f32x8 zero8 = {0.f, 0.f, 0.f, 0.f, 0.f, 0.f, 0.f, 0.f};
  f32x8 acc[NTILES];
#pragma unroll
  for (int t = 0; t < NTILES; ++t) acc[t] = zero8;

  const int arow = i0 + (lane & 15);
  const int ka   = (lane >> 4) * 8;
  const int kb   = (lane >> 4) * 16;
  const unsigned short* aptr = Ab + (long)arow * K;

  for (int k0 = 0; k0 < K; k0 += 32) {
    BFrag af;
    af.u[0] = *(const uint4*)(aptr + k0 + ka);
    af.u[1] = *(const uint4*)(aptr + k0 + ka + 16);
    if (k0 + 32 < K) __builtin_prefetch(aptr + k0 + 32 + ka, 0, 1);
#pragma unroll
    for (int t = 0; t < NTILES; ++t) {
      const unsigned short* bptr =
          Wb + (long)(n0 + t * 16 + (lane & 15)) * K + k0 + ka;
      BFrag bfrag;
      bfrag.u[0] = *(const uint4*)(bptr);
      bfrag.u[1] = *(const uint4*)(bptr + 16);
      acc[t] = __builtin_amdgcn_wmma_f32_16x16x32_bf16(
          false, af.v, false, bfrag.v, (short)0, acc[t], false, false);
    }
  }
#pragma unroll
  for (int t = 0; t < NTILES; ++t)
#pragma unroll
    for (int r = 0; r < 8; ++r)
      gemm_epilogue<EPI>(acc[t][r], i0 + r + ((lane >> 4) << 3),
                         n0 + t * 16 + (lane & 15), N, z, bias, residual, out, M);
}

// ---------------------------------------------------------------------------
// TDM/LDS-staged WMMA GEMM for the big dense layers.
// Block = 8 waves = 128 rows x 64 cols. The 64-row weight strip (Wt rows
// n0..n0+63, 32 K at a time) is DMA'd into LDS by the Tensor Data Mover
// (one tensor_load_to_lds per k-step, issued by wave 0, double-buffered,
// TENSORcnt-synced). TDM pad feature: +4 DWORDs per 16-DWORD row -> 80B LDS
// row stride (16B aligned, spreads ds_load_b128 bank accesses).
// ---------------------------------------------------------------------------
template <int EPI>
__global__ void gemm16_lds_kernel(const unsigned short* __restrict__ A,
                                  const unsigned short* __restrict__ Wt,
                                  int M, int N, int K,
                                  const float* __restrict__ bias,
                                  const float* __restrict__ residual,
                                  void* __restrict__ out) {
  constexpr int ROWB = 80;                     // padded LDS row stride (bytes)
  constexpr int BUFB = 64 * ROWB;              // one k-step buffer (5120 B)
  __shared__ __align__(16) unsigned char smem[2 * BUFB];

  const int lane = threadIdx.x & 31;
  const int w    = threadIdx.x >> 5;           // wave id in block (0..7)
  const int nstrips = N >> 6;
  const int ib = blockIdx.x / nstrips, jb = blockIdx.x % nstrips;
  const int i0 = ib * 128 + w * 16;
  const int n0 = jb * 64;

  // --- Tensor DMA descriptor (CDNA5 ISA ch.8) ---
  // group1: data_size=2B, pad_enable, pad_interval=16dw, pad_amount=4dw,
  //         tensor_dim0=K, tensor_dim1=N, tile_dim0=32, tile_dim1=64,
  //         tensor_dim0_stride=K
  u32x8 g1;
  g1[0] = (1u << 16) | (1u << 20) | (3u << 22) | (3u << 25);
  g1[1] = ((unsigned)K & 0xFFFFu) << 16;
  g1[2] = ((unsigned)K >> 16) | (((unsigned)N & 0xFFFFu) << 16);
  g1[3] = ((unsigned)N >> 16) | (32u << 16);
  g1[4] = 64u;
  g1[5] = (unsigned)K;
  g1[6] = 0u;
  g1[7] = 0u;

  const unsigned lds_base = (unsigned)(uintptr_t)&smem[0];   // low 32b = LDS offset
  const unsigned short* wbase = Wt + (long)n0 * K;

  auto tdm_load = [&](int kOff, unsigned ldsOff) {
    unsigned long long ga = (unsigned long long)(uintptr_t)(wbase + kOff);
    u32x4 g0;
    g0[0] = 1u;                                        // count=1, normal mode
    g0[1] = ldsOff;                                    // lds_addr
    g0[2] = (unsigned)ga;                              // global_addr[31:0]
    g0[3] = (unsigned)((ga >> 32) & 0x01FFFFFFu) | (2u << 30);   // addr[56:32], type=2
    asm volatile("tensor_load_to_lds %0, %1" :: "s"(g0), "s"(g1) : "memory");
  };

  // prologue: buffer 0 <- k0 = 0
  if (w == 0) {
    tdm_load(0, lds_base);
    __builtin_amdgcn_s_wait_tensorcnt(0);
  }
  __syncthreads();

  f32x8 zero8 = {0.f, 0.f, 0.f, 0.f, 0.f, 0.f, 0.f, 0.f};
  f32x8 acc[4] = {zero8, zero8, zero8, zero8};
  const int arow = i0 + (lane & 15);
  const int ka   = (lane >> 4) * 8;
  const int kbB  = (lane >> 4) * 16;            // byte offset of B chunk in LDS row
  const unsigned short* aptr = A + (long)arow * K;

  for (int k0 = 0; k0 < K; k0 += 32) {
    const int cur = (k0 >> 5) & 1;
    if (w == 0 && k0 + 32 < K)
      tdm_load(k0 + 32, lds_base + (cur ^ 1) * BUFB);   // prefetch next buffer

    BFrag af;
    af.u[0] = *(const uint4*)(aptr + k0 + ka);
    af.u[1] = *(const uint4*)(aptr + k0 + ka + 16);

    const unsigned char* bbuf = smem + cur * BUFB;
#pragma unroll
    for (int t = 0; t < 4; ++t) {
      const unsigned char* bp = bbuf + (t * 16 + (lane & 15)) * ROWB + kbB;
      BFrag bfrag;
      bfrag.u[0] = *(const uint4*)(bp);
      bfrag.u[1] = *(const uint4*)(bp + 32);
      acc[t] = __builtin_amdgcn_wmma_f32_16x16x32_bf16(
          false, af.v, false, bfrag.v, (short)0, acc[t], false, false);
    }

    if (w == 0 && k0 + 32 < K) __builtin_amdgcn_s_wait_tensorcnt(0);
    __syncthreads();                             // publish next buffer / guard reuse
  }

#pragma unroll
  for (int t = 0; t < 4; ++t)
#pragma unroll
    for (int r = 0; r < 8; ++r)
      gemm_epilogue<EPI>(acc[t][r], i0 + r + ((lane >> 4) << 3),
                         n0 + t * 16 + (lane & 15), N, 0, bias, residual, out, M);
}

// ---------------------------------------------------------------------------
// kr_small = rel_enc @ wkr + bkr   [201,512] fp32 + per-head bf16 copy (padded)
// ---------------------------------------------------------------------------
__global__ void krs_kernel(const float* __restrict__ rel_enc, const float* __restrict__ wkr,
                           const float* __restrict__ bkr,
                           float* __restrict__ krs, unsigned short* __restrict__ krs_h) {
  int idx = blockIdx.x * blockDim.x + threadIdx.x;
  if (idx >= kRP * kD) return;
  int r = idx / kD, n = idx % kD;
  float c = 0.f;
  if (r < kR) {
    c = bkr[n];
    for (int k = 0; k < kD; ++k) c += rel_enc[(long)r * kD + k] * wkr[(long)k * kD + n];
  }
  krs[idx] = c;
  int h = n >> 6, d = n & 63;
  krs_h[((long)h * kRP + r) * kDim + d] = f2bf(c);   // [H, RP, DIM], padded rows = 0
}

__global__ void bias1s_kernel(const float* __restrict__ krs, const float* __restrict__ wb1,
                              const float* __restrict__ bb1, float* __restrict__ b1s) {
  int idx = blockIdx.x * blockDim.x + threadIdx.x;
  if (idx >= kR * kH) return;
  int r = idx / kH, h = idx % kH;
  float c = bb1[h];
  for (int n = 0; n < kD; ++n) c += krs[(long)r * kD + n] * wb1[(long)n * kH + h];
  b1s[(long)r * kH + h] = c;
}

__global__ void bias0_kernel(const unsigned short* __restrict__ keh, const float* __restrict__ wb0,
                             const float* __restrict__ bb0, float* __restrict__ b0) {
  int idx = blockIdx.x * blockDim.x + threadIdx.x;    // [B,T,H']
  if (idx >= kBT * kH) return;
  int hp = idx % kH, bt = idx / kH, b = bt / kT, t = bt % kT;
  float c = bb0[hp];
  for (int h = 0; h < kH; ++h)
    for (int d = 0; d < kDim; ++d)
      c += bf2f(keh[(((long)(b * kH + h) * kT) + t) * kDim + d]) *
           wb0[(long)(h * kDim + d) * kH + hp];
  b0[idx] = c;
}

// ---------------------------------------------------------------------------
// scores = QK^T/8 + p[gather] + bias0 + bias1 ; causal & padding mask -> bf16
// ---------------------------------------------------------------------------
__global__ void scores_kernel(const unsigned short* __restrict__ qh,
                              const unsigned short* __restrict__ keh,
                              const float* __restrict__ p,       // [BH,T,RP]
                              const float* __restrict__ bias0,   // [B,T,H]
                              const float* __restrict__ b1s,     // [R,H]
                              const unsigned char* __restrict__ vmask, // [B,T]
                              unsigned short* __restrict__ scores) {   // [BH,T,T]
  const int lane = threadIdx.x & 31;
  const int wid  = (blockIdx.x * blockDim.x + threadIdx.x) >> 5;
  const int z = blockIdx.z, b = z / kH, h = z % kH;
  const int iw = wid / (kT / 64), jw = wid % (kT / 64);
  if (iw >= kT / 16) return;
  const int i0 = iw * 16, j0 = jw * 64;
  unsigned short* srow = scores + (long)z * kT * kT;

  if (j0 > i0 + 15) {            // strip entirely above the diagonal
    unsigned short neg = f2bf(-1e9f);
#pragma unroll
    for (int t = 0; t < 4; ++t)
#pragma unroll
      for (int r = 0; r < 8; ++r) {
        int i = i0 + r + ((lane >> 4) << 3), j = j0 + t * 16 + (lane & 15);
        srow[(long)i * kT + j] = neg;
      }
    return;
  }

  const unsigned short* Aq = qh + (long)z * kT * kDim;
  const unsigned short* Bk = keh + (long)z * kT * kDim;
  f32x8 zero8 = {0.f, 0.f, 0.f, 0.f, 0.f, 0.f, 0.f, 0.f};
  f32x8 acc[4] = {zero8, zero8, zero8, zero8};
  const int arow = i0 + (lane & 15);
  const int ka = (lane >> 4) * 8, kb = (lane >> 4) * 16;
  const unsigned short* aptr = Aq + (long)arow * kDim;

#pragma unroll
  for (int k0 = 0; k0 < kDim; k0 += 32) {
    BFrag af;
    af.u[0] = *(const uint4*)(aptr + k0 + ka);
    af.u[1] = *(const uint4*)(aptr + k0 + ka + 16);
#pragma unroll
    for (int t = 0; t < 4; ++t) {
      const unsigned short* bptr =
          Bk + (long)(j0 + t * 16 + (lane & 15)) * kDim + k0 + ka;
      BFrag bfrag;
      bfrag.u[0] = *(const uint4*)(bptr);
      bfrag.u[1] = *(const uint4*)(bptr + 16);
      acc[t] = __builtin_amdgcn_wmma_f32_16x16x32_bf16(
          false, af.v, false, bfrag.v, (short)0, acc[t], false, false);
    }
  }

#pragma unroll
  for (int t = 0; t < 4; ++t)
#pragma unroll
    for (int r = 0; r < 8; ++r) {
      int i = i0 + r + ((lane >> 4) << 3);
      int j = j0 + t * 16 + (lane & 15);
      int rel = j - i; rel = (rel < -kL ? -kL : (rel > kL ? kL : rel)) + kL;
      float c = acc[t][r] * 0.125f
              + p[((long)z * kT + i) * kRP + rel]
              + bias0[((long)b * kT + j) * kH + h]
              + b1s[(long)rel * kH + h];
      bool keep = (j <= i) && (vmask[b * kT + j] != 0);
      srow[(long)i * kT + j] = f2bf(keep ? c : -1e9f);
    }
}

// ---------------------------------------------------------------------------
// In-place softmax over rows of 512 (bf16). One wave per row.
// ---------------------------------------------------------------------------
__global__ void softmax_kernel(unsigned short* __restrict__ attn) {
  int row  = blockIdx.x * (blockDim.x >> 5) + (threadIdx.x >> 5);
  int lane = threadIdx.x & 31;
  if (row >= kBH * kT) return;
  unsigned short* r = attn + (long)row * kT;
  float v[16], mx = -3e38f;
#pragma unroll
  for (int i = 0; i < 16; ++i) { float t = bf2f(r[lane + 32 * i]); v[i] = t; mx = fmaxf(mx, t); }
#pragma unroll
  for (int m = 16; m >= 1; m >>= 1) mx = fmaxf(mx, __shfl_xor(mx, m, 32));
  float s = 0.f;
#pragma unroll
  for (int i = 0; i < 16; ++i) { float e = __expf(v[i] - mx); v[i] = e; s += e; }
#pragma unroll
  for (int m = 16; m >= 1; m >>= 1) s += __shfl_xor(s, m, 32);
  float inv = 1.0f / s;
#pragma unroll
  for (int i = 0; i < 16; ++i) r[lane + 32 * i] = f2bf(v[i] * inv);
}

// ---------------------------------------------------------------------------
// Host-side launch
// ---------------------------------------------------------------------------
static inline dim3 gemm_grid(int M, int N, int NT) {
  long waves = (long)(M / 16) * ((N / 16) / NT);
  return dim3((unsigned)((waves * 32 + 255) / 256), 1, 1);
}
static inline dim3 gemm_lds_grid(int M, int N) {
  return dim3((unsigned)((M / 128) * (N / 64)), 1, 1);
}

extern "C" void kernel_launch(void* const* d_in, const int* in_sizes, int n_in,
                              void* d_out, int out_size, void* d_ws, size_t ws_size,
                              hipStream_t stream) {
  (void)in_sizes; (void)n_in; (void)out_size; (void)ws_size;
  const float* values  = (const float*)d_in[0];
  const unsigned char* vmask = (const unsigned char*)d_in[1];
  const float* rel_enc = (const float*)d_in[2];
  const float* ln0_g = (const float*)d_in[3];
  const float* ln0_b = (const float*)d_in[4];
  const float* w_h0  = (const float*)d_in[5];
  const float* b_h0  = (const float*)d_in[6];
  const float* wq    = (const float*)d_in[7];
  const float* bq    = (const float*)d_in[8];
  const float* wke   = (const float*)d_in[9];
  const float* bke   = (const float*)d_in[10];
  const float* wkv   = (const float*)d_in[11];
  const float* bkv   = (const float*)d_in[12];
  const float* wkr   = (const float*)d_in[13];
  const float* bkr   = (const float*)d_in[14];
  const float* wb0   = (const float*)d_in[15];
  const float* bb0   = (const float*)d_in[16];
  const float* wb1   = (const float*)d_in[17];
  const float* bb1   = (const float*)d_in[18];
  const float* ln1_g = (const float*)d_in[19];
  const float* ln1_b = (const float*)d_in[20];
  const float* w_h1  = (const float*)d_in[21];
  const float* b_h1  = (const float*)d_in[22];
  const float* w_o1  = (const float*)d_in[23];
  const float* b_o1  = (const float*)d_in[24];
  float* out = (float*)d_out;

  // Workspace carve-up (256B aligned)
  char* w = (char*)d_ws;
  size_t off = 0;
  auto take = [&](size_t bytes) {
    off = (off + 255) & ~(size_t)255;
    char* q = w + off; off += bytes; return q;
  };
  unsigned short* wt_h0 = (unsigned short*)take((size_t)kD * kHid * 2);
  unsigned short* wt_q  = (unsigned short*)take((size_t)kHid * kD * 2);
  unsigned short* wt_ke = (unsigned short*)take((size_t)kHid * kD * 2);
  unsigned short* wt_kv = (unsigned short*)take((size_t)kHid * kD * 2);
  unsigned short* wt_h1 = (unsigned short*)take((size_t)kD * kHid * 2);
  unsigned short* wt_o1 = (unsigned short*)take((size_t)kHid * kD * 2);
  unsigned short* xln   = (unsigned short*)take((size_t)kBT * kD * 2);
  unsigned short* X1    = (unsigned short*)take((size_t)kBT * kHid * 2);
  unsigned short* qh    = (unsigned short*)take((size_t)kBH * kT * kDim * 2);
  unsigned short* keh   = (unsigned short*)take((size_t)kBH * kT * kDim * 2);
  unsigned short* kvT   = (unsigned short*)take((size_t)kBH * kDim * kT * 2);
  float*          krs   = (float*)take((size_t)kRP * kD * 4);
  unsigned short* krs_h = (unsigned short*)take((size_t)kH * kRP * kDim * 2);
  float*          b1s   = (float*)take((size_t)kRP * kH * 4);
  float*          b0    = (float*)take((size_t)kBT * kH * 4);
  float*          ptab  = (float*)take((size_t)kBH * kT * kRP * 4);
  unsigned short* attn  = (unsigned short*)take((size_t)kBH * kT * kT * 2);
  float*          y1    = (float*)take((size_t)kBT * kD * 4);

  const int TPB = 256;

  // 1) weights -> bf16, transposed to [N,K]
  auto cvt = [&](const float* src, unsigned short* dst, int K, int N) {
    long n = (long)K * N;
    cvtT_kernel<<<dim3((unsigned)((n + TPB - 1) / TPB)), TPB, 0, stream>>>(src, dst, K, N);
  };
  cvt(w_h0, wt_h0, kD, kHid);
  cvt(wq,   wt_q,  kHid, kD);
  cvt(wke,  wt_ke, kHid, kD);
  cvt(wkv,  wt_kv, kHid, kD);
  cvt(w_h1, wt_h1, kD, kHid);
  cvt(w_o1, wt_o1, kHid, kD);

  // 2) LN0 -> bf16
  ln_kernel<<<dim3(kBT / 8), TPB, 0, stream>>>(values, ln0_g, ln0_b, xln);

  // 3) X1 = relu(xln @ w_h0 + b_h0)   [4096 x 2048, K=512]  (TDM-staged)
  gemm16_lds_kernel<EPI_RELU_BF16><<<gemm_lds_grid(kBT, kHid), TPB, 0, stream>>>(
      xln, wt_h0, kBT, kHid, kD, b_h0, nullptr, X1);

  // 4) q/ke -> [B,H,T,DIM], kv -> [B,H,DIM,T]   [4096 x 512, K=2048]  (TDM-staged)
  gemm16_lds_kernel<EPI_HEADS><<<gemm_lds_grid(kBT, kD), TPB, 0, stream>>>(
      X1, wt_q, kBT, kD, kHid, bq, nullptr, qh);
  gemm16_lds_kernel<EPI_HEADS><<<gemm_lds_grid(kBT, kD), TPB, 0, stream>>>(
      X1, wt_ke, kBT, kD, kHid, bke, nullptr, keh);
  gemm16_lds_kernel<EPI_KVT><<<gemm_lds_grid(kBT, kD), TPB, 0, stream>>>(
      X1, wt_kv, kBT, kD, kHid, bkv, nullptr, kvT);

  // 5) kr_small, bias1_small, bias0
  krs_kernel<<<dim3((kRP * kD + TPB - 1) / TPB), TPB, 0, stream>>>(rel_enc, wkr, bkr, krs, krs_h);
  bias1s_kernel<<<dim3((kR * kH + TPB - 1) / TPB), TPB, 0, stream>>>(krs, wb1, bb1, b1s);
  bias0_kernel<<<dim3((kBT * kH + TPB - 1) / TPB), TPB, 0, stream>>>(keh, wb0, bb0, b0);

  // 6) p[b,h,i,r] = q . kr_small  per head   [T x 208, K=64], batched over BH
  {
    dim3 g = gemm_grid(kT, kRP, 1); g.z = kBH;
    gemm16_kernel<EPI_F32, 1><<<g, TPB, 0, stream>>>(
        qh, krs_h, kT, kRP, kDim,
        (long)kT * kDim, (long)kRP * kDim, kH, nullptr, nullptr, ptab);
  }

  // 7) scores (fused bias/mask) -> bf16
  {
    dim3 g(32, 1, kBH);   // 256 waves per (b,h)
    scores_kernel<<<g, TPB, 0, stream>>>(qh, keh, ptab, b0, b1s, vmask, attn);
  }

  // 8) softmax in place
  softmax_kernel<<<dim3(kBH * kT / 8), TPB, 0, stream>>>(attn);

  // 9) y1 = values + attn @ kv    [T x 64, K=512], batched over BH
  {
    dim3 g = gemm_grid(kT, kDim, 4); g.z = kBH;
    gemm16_kernel<EPI_ATTNOUT, 4><<<g, TPB, 0, stream>>>(
        attn, kvT, kT, kDim, kT,
        (long)kT * kT, (long)kDim * kT, 0, nullptr, values, y1);
  }

  // 10) block1: LN1 -> relu dense -> dense + residual  (TDM-staged)
  ln_kernel<<<dim3(kBT / 8), TPB, 0, stream>>>(y1, ln1_g, ln1_b, xln);
  gemm16_lds_kernel<EPI_RELU_BF16><<<gemm_lds_grid(kBT, kHid), TPB, 0, stream>>>(
      xln, wt_h1, kBT, kHid, kD, b_h1, nullptr, X1);
  gemm16_lds_kernel<EPI_OUT_RES><<<gemm_lds_grid(kBT, kD), TPB, 0, stream>>>(
      X1, wt_o1, kBT, kD, kHid, b_o1, y1, out);
}
